// Attention_90220083019846
// MI455X (gfx1250) — compile-verified, matrix operand order checked by reading, not run on hardware
//
#include <hip/hip_runtime.h>
#include <hip/hip_bf16.h>
#include <math.h>

typedef __attribute__((ext_vector_type(16))) _Float16 v16h;
typedef __attribute__((ext_vector_type(8)))  _Float16 v8h;
typedef __attribute__((ext_vector_type(8)))  float    v8f;

#define B_  2
#define L_  2048
#define D_  1024
#define H_  16
#define HD_ 64

// ---------------------------------------------------------------------------
// WMMA fragment loader (16-bit, 16x32 A / 32x16 B, K-major per lane).
// Per-lane packing (ISA 7.12.2): element e -> K = (e>=8?16:0) + (lane>>4)*8 + (e&7)
// => two contiguous 8-half (16B) runs at k0+(lane>>4)*8 and +16.
// Caller passes the per-lane row (already includes lane&15).
// ---------------------------------------------------------------------------
__device__ __forceinline__ v16h load_frag16(const _Float16* __restrict__ base,
                                            int row, int ld, int k0) {
    int lane = threadIdx.x;                    // blockDim.x == 32 in all users
    const _Float16* p = base + (size_t)row * ld + k0 + ((lane >> 4) << 3);
    v8h lo = *reinterpret_cast<const v8h*>(p);
    v8h hi = *reinterpret_cast<const v8h*>(p + 16);
    return __builtin_shufflevector(lo, hi, 0,1,2,3,4,5,6,7,8,9,10,11,12,13,14,15);
}

__device__ __forceinline__ v8f wmma_f16(v16h a, v16h b, v8f c) {
    return __builtin_amdgcn_wmma_f32_16x16x32_f16(false, a, false, b,
                                                  (short)0, c, false, false);
}

// ---------------------------------------------------------------------------
// f32 -> f16 elementwise
// ---------------------------------------------------------------------------
__global__ __launch_bounds__(256) void cvt_f16(const float* __restrict__ in,
                                               _Float16* __restrict__ out, int n) {
    int i = blockIdx.x * 256 + threadIdx.x;
    if (i < n) out[i] = (_Float16)in[i];
}

// f32 (1024x1024) -> f16 transposed
__global__ __launch_bounds__(256) void cvt_f16_t(const float* __restrict__ in,
                                                 _Float16* __restrict__ out) {
    int c = blockIdx.x * 16 + threadIdx.x;
    int r = blockIdx.y * 16 + threadIdx.y;
    out[(size_t)c * D_ + r] = (_Float16)in[(size_t)r * D_ + c];
}

// ---------------------------------------------------------------------------
// Projection GEMM: out = Xh(4096x1024) @ W (via Wt, K-contiguous rows).
// One wave per 16x64 tile: A-frag loaded once per K-step, reused for 4 B-frags.
// vtrans=0: head layout [bh][i][64]; vtrans=1: [bh][64][i].
// ---------------------------------------------------------------------------
__global__ __launch_bounds__(256) void gemm_proj(const _Float16* __restrict__ Xh,
                                                 const _Float16* __restrict__ Wt,
                                                 _Float16* __restrict__ out,
                                                 int vtrans) {
    int lane   = threadIdx.x;
    int tid    = blockIdx.x * 8 + threadIdx.y;
    int tile_r = tid >> 4;                     // 256 row tiles (16 rows each)
    int tile_c = tid & 15;                     // 16 col tiles (64 cols each)
    int arow   = tile_r * 16 + (lane & 15);
    int bbase  = tile_c * 64 + (lane & 15);
    v8f acc0 = {}, acc1 = {}, acc2 = {}, acc3 = {};
    for (int k0 = 0; k0 < D_; k0 += 32) {
        v16h a = load_frag16(Xh, arow, D_, k0);
        acc0 = wmma_f16(a, load_frag16(Wt, bbase +  0, D_, k0), acc0);
        acc1 = wmma_f16(a, load_frag16(Wt, bbase + 16, D_, k0), acc1);
        acc2 = wmma_f16(a, load_frag16(Wt, bbase + 32, D_, k0), acc2);
        acc3 = wmma_f16(a, load_frag16(Wt, bbase + 48, D_, k0), acc3);
    }
    int hl = lane >> 4, n = lane & 15;
    #pragma unroll
    for (int c = 0; c < 4; ++c) {
        v8f acc = (c == 0) ? acc0 : (c == 1) ? acc1 : (c == 2) ? acc2 : acc3;
        #pragma unroll
        for (int r = 0; r < 8; ++r) {
            int ri = tile_r * 16 + r + hl * 8;     // global token row
            int d  = tile_c * 64 + c * 16 + n;
            int bI = ri >> 11, i = ri & (L_ - 1);
            int h  = d >> 6,  dh = d & 63;
            _Float16 val = (_Float16)acc[r];
            if (!vtrans)
                out[(((size_t)(bI * H_ + h)) * L_ + i) * HD_ + dh] = val;
            else
                out[(((size_t)(bI * H_ + h)) * HD_ + dh) * L_ + i] = val;
        }
    }
}

// ---------------------------------------------------------------------------
// A_c[b,h,j] = sum_d cb[h*64+d] * K[b,h,j,d]
// ---------------------------------------------------------------------------
__global__ __launch_bounds__(256) void bias_ck(const _Float16* __restrict__ Kh,
                                               const float* __restrict__ cb,
                                               float* __restrict__ bias) {
    int t  = blockIdx.x * 256 + threadIdx.x;   // < B*H*L = 65536
    int bh = t >> 11;
    int j  = t & (L_ - 1);
    int h  = bh & (H_ - 1);
    const _Float16* kr = Kh + ((size_t)bh * L_ + j) * HD_;
    const float*    cc = cb + h * HD_;
    float s = 0.f;
    #pragma unroll
    for (int d = 0; d < HD_; ++d) s += cc[d] * (float)kr[d];
    bias[t] = s;
}

// ---------------------------------------------------------------------------
// Flash attention: one wave per (b,h, 16-query tile). 32 keys per iteration.
// S via 4 WMMAs, online softmax, P staged f16 in LDS, PV via 4 WMMAs.
// ---------------------------------------------------------------------------
__global__ __launch_bounds__(256) void attn(const _Float16* __restrict__ Qh,
                                            const _Float16* __restrict__ Kh,
                                            const _Float16* __restrict__ Vt,
                                            const float*    __restrict__ biasA,
                                            _Float16*       __restrict__ Of) {
    __shared__ __align__(16) _Float16 ldsP[8][16 * 32];
    int lane = threadIdx.x, wave = threadIdx.y;
    int tid  = blockIdx.x * 8 + wave;
    int bh   = tid >> 7;                       // 32 (b,h) pairs
    int it   = tid & 127;                      // 128 query tiles
    int hl = lane >> 4, n = lane & 15;
    int i0 = it * 16;

    const _Float16* Qb = Qh + (size_t)bh * L_ * HD_;
    const _Float16* Kb = Kh + (size_t)bh * L_ * HD_;
    const _Float16* Vb = Vt + (size_t)bh * HD_ * L_;
    const float*    bb = biasA + (size_t)bh * L_;

    v16h qa0 = load_frag16(Qb, i0 + n, HD_, 0);
    v16h qa1 = load_frag16(Qb, i0 + n, HD_, 32);

    float m_st[8], l_st[8];
    v8f o0 = {}, o1 = {}, o2 = {}, o3 = {};
    #pragma unroll
    for (int r = 0; r < 8; ++r) { m_st[r] = -3.0e38f; l_st[r] = 0.f; }

    for (int j0 = 0; j0 < L_; j0 += 32) {
        v8f c0 = {}, c1 = {};
        c0 = wmma_f16(qa0, load_frag16(Kb, j0 + n,      HD_, 0),  c0);
        c0 = wmma_f16(qa1, load_frag16(Kb, j0 + n,      HD_, 32), c0);
        c1 = wmma_f16(qa0, load_frag16(Kb, j0 + 16 + n, HD_, 0),  c1);
        c1 = wmma_f16(qa1, load_frag16(Kb, j0 + 16 + n, HD_, 32), c1);
        float b0 = bb[j0 + n];
        float b1 = bb[j0 + 16 + n];
        #pragma unroll
        for (int r = 0; r < 8; ++r) {
            float s0 = c0[r] + b0;
            float s1 = c1[r] + b1;
            float v = fmaxf(s0, s1);
            #pragma unroll
            for (int off = 8; off; off >>= 1) v = fmaxf(v, __shfl_xor(v, off, 32));
            float mn   = fmaxf(m_st[r], v);
            float corr = __expf(m_st[r] - mn);
            float p0   = __expf(s0 - mn);
            float p1   = __expf(s1 - mn);
            float sum  = p0 + p1;
            #pragma unroll
            for (int off = 8; off; off >>= 1) sum += __shfl_xor(sum, off, 32);
            l_st[r] = l_st[r] * corr + sum;
            m_st[r] = mn;
            o0[r] *= corr; o1[r] *= corr; o2[r] *= corr; o3[r] *= corr;
            int m = r + hl * 8;
            ldsP[wave][m * 32 + n]      = (_Float16)p0;
            ldsP[wave][m * 32 + 16 + n] = (_Float16)p1;
        }
        asm volatile("s_wait_dscnt 0x0" ::: "memory");   // cross-lane LDS hazard
        v16h pa = load_frag16(&ldsP[wave][0], n, 32, 0); // P tile 16x32, A-frag
        o0 = wmma_f16(pa, load_frag16(Vb,  0 + n, L_, j0), o0);
        o1 = wmma_f16(pa, load_frag16(Vb, 16 + n, L_, j0), o1);
        o2 = wmma_f16(pa, load_frag16(Vb, 32 + n, L_, j0), o2);
        o3 = wmma_f16(pa, load_frag16(Vb, 48 + n, L_, j0), o3);
    }

    int bI = bh >> 4, h = bh & (H_ - 1);
    #pragma unroll
    for (int r = 0; r < 8; ++r) {
        int i = i0 + r + hl * 8;
        float inv = 1.0f / l_st[r];
        size_t ro = ((size_t)(bI * L_ + i)) * D_ + h * HD_;
        Of[ro +  0 + n] = (_Float16)(o0[r] * inv);
        Of[ro + 16 + n] = (_Float16)(o1[r] * inv);
        Of[ro + 32 + n] = (_Float16)(o2[r] * inv);
        Of[ro + 48 + n] = (_Float16)(o3[r] * inv);
    }
}

// ---------------------------------------------------------------------------
// Z = X + Of @ Wo_w^T + Wo_b   (f32 out; Wo_w rows are already K-contiguous)
// One wave per 16x64 tile, A-frag reused across 4 B-frags.
// ---------------------------------------------------------------------------
__global__ __launch_bounds__(256) void gemm_out(const _Float16* __restrict__ Of,
                                                const _Float16* __restrict__ WoH,
                                                const float* __restrict__ X,
                                                const float* __restrict__ Wob,
                                                float* __restrict__ Z) {
    int lane   = threadIdx.x;
    int tid    = blockIdx.x * 8 + threadIdx.y;
    int tile_r = tid >> 4;                     // 256 row tiles
    int tile_c = tid & 15;                     // 16 col tiles (64 cols)
    int arow   = tile_r * 16 + (lane & 15);
    int bbase  = tile_c * 64 + (lane & 15);
    v8f acc0 = {}, acc1 = {}, acc2 = {}, acc3 = {};
    for (int k0 = 0; k0 < D_; k0 += 32) {
        v16h a = load_frag16(Of, arow, D_, k0);
        acc0 = wmma_f16(a, load_frag16(WoH, bbase +  0, D_, k0), acc0);
        acc1 = wmma_f16(a, load_frag16(WoH, bbase + 16, D_, k0), acc1);
        acc2 = wmma_f16(a, load_frag16(WoH, bbase + 32, D_, k0), acc2);
        acc3 = wmma_f16(a, load_frag16(WoH, bbase + 48, D_, k0), acc3);
    }
    int hl = lane >> 4, n = lane & 15;
    #pragma unroll
    for (int c = 0; c < 4; ++c) {
        v8f acc = (c == 0) ? acc0 : (c == 1) ? acc1 : (c == 2) ? acc2 : acc3;
        #pragma unroll
        for (int r = 0; r < 8; ++r) {
            int ri = tile_r * 16 + r + hl * 8;
            int d  = tile_c * 64 + c * 16 + n;
            size_t idx = (size_t)ri * D_ + d;
            Z[idx] = acc[r] + X[idx] + Wob[d];
        }
    }
}

// ---------------------------------------------------------------------------
// In-place LayerNorm over last dim (1024), one block per row.
// ---------------------------------------------------------------------------
__global__ __launch_bounds__(256) void layernorm(float* __restrict__ Z,
                                                 const float* __restrict__ g,
                                                 const float* __restrict__ bta) {
    __shared__ float red0[8], red1[8], stats[2];
    float* zr = Z + (size_t)blockIdx.x * D_;
    float v[4], s = 0.f, s2 = 0.f;
    #pragma unroll
    for (int k = 0; k < 4; ++k) {
        v[k] = zr[threadIdx.x + 256 * k];
        s += v[k]; s2 += v[k] * v[k];
    }
    #pragma unroll
    for (int off = 16; off; off >>= 1) {
        s  += __shfl_xor(s,  off, 32);
        s2 += __shfl_xor(s2, off, 32);
    }
    int wid = threadIdx.x >> 5;
    if ((threadIdx.x & 31) == 0) { red0[wid] = s; red1[wid] = s2; }
    __syncthreads();
    if (threadIdx.x == 0) {
        float ts = 0.f, ts2 = 0.f;
        #pragma unroll
        for (int i = 0; i < 8; ++i) { ts += red0[i]; ts2 += red1[i]; }
        float mu  = ts / (float)D_;
        float var = ts2 / (float)D_ - mu * mu;
        stats[0] = mu;
        stats[1] = rsqrtf(var + 1e-5f);
    }
    __syncthreads();
    float mu = stats[0], rs = stats[1];
    #pragma unroll
    for (int k = 0; k < 4; ++k) {
        int j = threadIdx.x + 256 * k;
        zr[j] = (v[k] - mu) * rs * g[j] + bta[j];
    }
}

// ---------------------------------------------------------------------------
extern "C" void kernel_launch(void* const* d_in, const int* in_sizes, int n_in,
                              void* d_out, int out_size, void* d_ws, size_t ws_size,
                              hipStream_t stream) {
    const float* X    = (const float*)d_in[0];
    // d_in[1] (Y) is unused by the reference
    const float* Wq   = (const float*)d_in[2];
    const float* Wk   = (const float*)d_in[3];
    const float* Wv   = (const float*)d_in[4];
    const float* cb   = (const float*)d_in[5];
    const float* Wo_w = (const float*)d_in[6];
    const float* Wo_b = (const float*)d_in[7];
    const float* ln_g = (const float*)d_in[8];
    const float* ln_b = (const float*)d_in[9];
    float* Z = (float*)d_out;

    const size_t MB = 1u << 20;
    char* ws = (char*)d_ws;
    _Float16* Xh  = (_Float16*)(ws);            //  8 MB  (4096x1024)
    _Float16* WqT = (_Float16*)(ws +  8 * MB);  //  2 MB  transposed
    _Float16* WkT = (_Float16*)(ws + 10 * MB);  //  2 MB
    _Float16* WvT = (_Float16*)(ws + 12 * MB);  //  2 MB
    _Float16* WoH = (_Float16*)(ws + 14 * MB);  //  2 MB  (row-major)
    _Float16* Qh  = (_Float16*)(ws + 16 * MB);  //  8 MB  [bh][i][64]
    _Float16* Kh  = (_Float16*)(ws + 24 * MB);  //  8 MB  [bh][i][64]
    _Float16* Vt  = (_Float16*)(ws + 32 * MB);  //  8 MB  [bh][64][i]
    _Float16* Of  = (_Float16*)(ws + 40 * MB);  //  8 MB  [ri][1024]
    float*  biasA = (float*)   (ws + 48 * MB);  // 256 KB [bh][j]

    dim3 wgGemm(32, 8);                         // 8 waves/block

    cvt_f16  <<<(B_ * L_ * D_) / 256, 256, 0, stream>>>(X, Xh, B_ * L_ * D_);
    cvt_f16_t<<<dim3(64, 64), dim3(16, 16), 0, stream>>>(Wq, WqT);
    cvt_f16_t<<<dim3(64, 64), dim3(16, 16), 0, stream>>>(Wk, WkT);
    cvt_f16_t<<<dim3(64, 64), dim3(16, 16), 0, stream>>>(Wv, WvT);
    cvt_f16  <<<(D_ * D_) / 256, 256, 0, stream>>>(Wo_w, WoH, D_ * D_);

    // 256 row-tiles x 16 col-tiles = 4096 waves = 512 blocks
    gemm_proj<<<512, wgGemm, 0, stream>>>(Xh, WqT, Qh, 0);
    gemm_proj<<<512, wgGemm, 0, stream>>>(Xh, WkT, Kh, 0);
    gemm_proj<<<512, wgGemm, 0, stream>>>(Xh, WvT, Vt, 1);

    bias_ck<<<(B_ * H_ * L_) / 256, 256, 0, stream>>>(Kh, cb, biasA);

    attn<<<(B_ * H_ * (L_ / 16)) / 8, wgGemm, 0, stream>>>(Qh, Kh, Vt, biasA, Of);

    gemm_out<<<512, wgGemm, 0, stream>>>(Of, WoH, X, Wo_b, Z);

    layernorm<<<B_ * L_, 256, 0, stream>>>(Z, ln_g, ln_b);
}
